// LSNNCell_3882650436359
// MI455X (gfx1250) — matrix operand order repaired and checked by reading, not compiled, not run on hardware
//
#include <hip/hip_runtime.h>

// ---------------------------------------------------------------------------
// LSNN cell, fused:  i_new = i_decayed + [x | z] @ [Wi | Wr]^T  (bf16 WMMA,
// f32 accumulate), with all elementwise LSNN state updates in the epilogue.
// ---------------------------------------------------------------------------

typedef __bf16 bf16_t;
typedef __attribute__((ext_vector_type(8)))  __bf16 v8bf;
typedef __attribute__((ext_vector_type(16))) __bf16 v16bf;
typedef __attribute__((ext_vector_type(8)))  float  v8f;
typedef __attribute__((ext_vector_type(4)))  float  v4f;

#define B_DIM   2048
#define N_INP   4096
#define N_OUT   4096
#define K_TOTAL (N_INP + N_OUT)   // 8192, concatenated reduction dim

#define BLK_M   128
#define BLK_N   128
#define KT      32
#define KTILES  (K_TOTAL / KT)    // 256
#define STRIDE  40                // padded bf16 row stride (32 + 8) -> 80B, 16B aligned
#define THREADS 256               // 8 waves (wave32)

__launch_bounds__(THREADS, 1)
__global__ void lsnn_fused_kernel(
    const float* __restrict__ x,    // (B, N_INP)
    const float* __restrict__ zin,  // (B, N_OUT)
    const float* __restrict__ vin,  // (B, N_OUT)
    const float* __restrict__ iin,  // (B, N_OUT)
    const float* __restrict__ bin,  // (B, N_OUT)
    const float* __restrict__ Wi,   // (N_OUT, N_INP)
    const float* __restrict__ Wr,   // (N_OUT, N_OUT)
    float* __restrict__ out)        // 4 x (B, N_OUT): z, v, i, b
{
  __shared__ bf16_t Xs[2][BLK_M * STRIDE];   // activations tile (bf16)
  __shared__ bf16_t Ws[2][BLK_N * STRIDE];   // weights tile (bf16)

  const int tid      = threadIdx.x;
  const int lane     = tid & 31;
  const int wave     = tid >> 5;
  const int wm       = wave & 3;     // 4 waves along M (32 rows each)
  const int wn       = wave >> 2;    // 2 waves along N (64 cols each)
  const int laneCol  = lane & 15;
  const int laneHalf = lane >> 4;

  const int nBase = blockIdx.x * BLK_N;
  const int mBase = blockIdx.y * BLK_M;

  // Staging: each thread fetches 16 consecutive k of one row (X and W)
  const int srow = tid >> 1;          // 0..127
  const int skof = (tid & 1) << 4;    // 0 or 16

  v4f xr[4], wr[4];

  auto stage_load = [&](int kt) {
    const int kk = kt * KT + skof;
    const float* xs;
    const float* ws;
    if (kk < N_INP) {              // phase 1: input_tensor @ Wi^T
      xs = x  + (size_t)(mBase + srow) * N_INP + kk;
      ws = Wi + (size_t)(nBase + srow) * N_INP + kk;
    } else {                       // phase 2: z @ Wr^T
      xs = zin + (size_t)(mBase + srow) * N_OUT + (kk - N_INP);
      ws = Wr  + (size_t)(nBase + srow) * N_OUT + (kk - N_INP);
    }
#pragma unroll
    for (int u = 0; u < 4; ++u) {
      xr[u] = *(const v4f*)(xs + u * 4);
      wr[u] = *(const v4f*)(ws + u * 4);
    }
  };

  auto stage_store = [&](int buf) {
    bf16_t xb[16], wb[16];
#pragma unroll
    for (int u = 0; u < 4; ++u)
#pragma unroll
      for (int e = 0; e < 4; ++e) {
        xb[u * 4 + e] = (bf16_t)xr[u][e];
        wb[u * 4 + e] = (bf16_t)wr[u][e];
      }
    bf16_t* xd = &Xs[buf][srow * STRIDE + skof];
    bf16_t* wd = &Ws[buf][srow * STRIDE + skof];
    *(v8bf*)(xd)     = *(const v8bf*)&xb[0];
    *(v8bf*)(xd + 8) = *(const v8bf*)&xb[8];
    *(v8bf*)(wd)     = *(const v8bf*)&wb[0];
    *(v8bf*)(wd + 8) = *(const v8bf*)&wb[8];
  };

  v8f acc[2][4];
  v8f zero;
#pragma unroll
  for (int h = 0; h < 8; ++h) zero[h] = 0.0f;
#pragma unroll
  for (int fm = 0; fm < 2; ++fm)
#pragma unroll
    for (int fn = 0; fn < 4; ++fn) acc[fm][fn] = zero;

  // Prologue: fill buffer 0
  stage_load(0);
  stage_store(0);
  __syncthreads();

  for (int kt = 0; kt < KTILES; ++kt) {
    const int cur = kt & 1;
    if (kt + 1 < KTILES) stage_load(kt + 1);   // global loads overlap WMMA

    // A fragments: 16x32 bf16, lane L -> row L%16,
    //   halves 0..7  = K {0..7}  (lanes 0-15) / {8..15}  (lanes 16-31)
    //   halves 8..15 = K {16..23}(lanes 0-15) / {24..31} (lanes 16-31)
    v16bf afrag[2];
#pragma unroll
    for (int fm = 0; fm < 2; ++fm) {
      const bf16_t* ap =
          &Xs[cur][(wm * 32 + fm * 16 + laneCol) * STRIDE + laneHalf * 8];
      v8bf lo = *(const v8bf*)(ap);
      v8bf hi = *(const v8bf*)(ap + 16);
#pragma unroll
      for (int h = 0; h < 8; ++h) { afrag[fm][h] = lo[h]; afrag[fm][h + 8] = hi[h]; }
    }

    // B fragments: 32x16 bf16, lane L -> column (W row) L%16,
    //   same K order as A: {0..7,16..23} for lanes 0-15, {8..15,24..31} for lanes 16-31
    v16bf bfrag[4];
#pragma unroll
    for (int fn = 0; fn < 4; ++fn) {
      const bf16_t* bp =
          &Ws[cur][(wn * 64 + fn * 16 + laneCol) * STRIDE + laneHalf * 8];
      v8bf lo = *(const v8bf*)(bp);
      v8bf hi = *(const v8bf*)(bp + 16);
#pragma unroll
      for (int h = 0; h < 8; ++h) { bfrag[fn][h] = lo[h]; bfrag[fn][h + 8] = hi[h]; }
    }

#pragma unroll
    for (int fm = 0; fm < 2; ++fm)
#pragma unroll
      for (int fn = 0; fn < 4; ++fn)
        acc[fm][fn] = __builtin_amdgcn_wmma_f32_16x16x32_bf16(
            false, afrag[fm], false, bfrag[fn],
            (short)0, acc[fm][fn], false, false);

    if (kt + 1 < KTILES) stage_store(cur ^ 1);  // fill other buffer
    __syncthreads();
  }

  // -------------------------------------------------------------------------
  // Epilogue: fused LSNN elementwise update.
  //   v_dec = v + 0.1*(i - v)            (DT*TAU_MEM_INV = 0.1, V_LEAK = 0)
  //   i_dec = 0.8*i                      (1 - DT*TAU_SYN_INV)
  //   b_dec = b + (DT/700)*(1 - b)
  //   z_new = heaviside(v_dec - b_dec)
  //   v_new = (1-z)*v_dec                (V_RESET = 0)
  //   i_new = i_dec + GEMM
  //   b_new = b_dec + z * (BETA/700)
  // C/D layout: VGPR r, lane L -> row (r + 8*(L/16)), col L%16.
  // -------------------------------------------------------------------------
  const size_t TOT = (size_t)B_DIM * N_OUT;
  float* zout = out;
  float* vout = out + TOT;
  float* iout = out + 2 * TOT;
  float* bout = out + 3 * TOT;

  const float k_vdec = 0.1f;
  const float k_idec = 0.8f;
  const float k_bdec = 1.4285714285714286e-6f;   // DT * TAU_ADAPT_INV
  const float k_binc = 2.5714285714285713e-3f;   // TAU_ADAPT_INV * BETA

#pragma unroll
  for (int fm = 0; fm < 2; ++fm) {
#pragma unroll
    for (int fn = 0; fn < 4; ++fn) {
      const int col  = nBase + wn * 64 + fn * 16 + laneCol;
      const int row0 = mBase + wm * 32 + fm * 16 + laneHalf * 8;
#pragma unroll
      for (int r = 0; r < 8; ++r) {
        const size_t idx = (size_t)(row0 + r) * N_OUT + col;
        const float a    = acc[fm][fn][r];
        const float vold = vin[idx];
        const float iold = iin[idx];
        const float bold = bin[idx];
        const float vdec = vold + k_vdec * (iold - vold);
        const float idec = k_idec * iold;
        const float bdec = bold + k_bdec * (1.0f - bold);
        const float z    = (vdec - bdec) > 0.0f ? 1.0f : 0.0f;
        zout[idx] = z;
        vout[idx] = (1.0f - z) * vdec;
        iout[idx] = idec + a;
        bout[idx] = bdec + z * k_binc;
      }
    }
  }
}

extern "C" void kernel_launch(void* const* d_in, const int* in_sizes, int n_in,
                              void* d_out, int out_size, void* d_ws, size_t ws_size,
                              hipStream_t stream) {
  const float* x   = (const float*)d_in[0];  // input_tensor (B, N_INP)
  const float* z   = (const float*)d_in[1];  // z            (B, N_OUT)
  const float* v   = (const float*)d_in[2];  // v            (B, N_OUT)
  const float* i_  = (const float*)d_in[3];  // i            (B, N_OUT)
  const float* b   = (const float*)d_in[4];  // b            (B, N_OUT)
  const float* Wi  = (const float*)d_in[5];  // input_weights     (N_OUT, N_INP)
  const float* Wr  = (const float*)d_in[6];  // recurrent_weights (N_OUT, N_OUT)
  float* out = (float*)d_out;

  dim3 grid(N_OUT / BLK_N, B_DIM / BLK_M);   // (32, 16)
  lsnn_fused_kernel<<<grid, THREADS, 0, stream>>>(x, z, v, i_, b, Wi, Wr, out);
}
